// PointerNetwork_20633022890689
// MI455X (gfx1250) — matrix operand order repaired and checked by reading, not compiled, not run on hardware
//
#include <hip/hip_runtime.h>

// ---------------------------------------------------------------------------
// Pointer network for MI455X (gfx1250, wave32, WMMA 16x16x32 f16 -> f32).
//
//   1. pack weights (f32 [N,K]) into WMMA B-fragment layout (f16)
//   2. embedding lookup -> x_seq f16 [S,B,E]
//   3. per encoder layer/dir: GX = X @ Wi^T + b (big WMMA GEMM, hoisted out of
//      the scan, A staged via double-buffered async global->LDS), then a
//      persistent scan kernel (1 WG per 16 batch rows, h/c state in LDS):
//      per step async-stage GX[s] into LDS, overlap with h @ Wh^T WMMA, then
//      pointwise LSTM cell. Gate layout [i,f,g,o].
//   4. enc_energy = (of1+ob1) @ w1^T + b1 (WMMA GEMM, f16, L2 resident)
//   5. persistent decoder: d0 cell (const input term hoisted), d1 cell,
//      de = h1 @ w2^T, tanh-attention + softmax -> out[B,S,S].
// ---------------------------------------------------------------------------

typedef __attribute__((ext_vector_type(8)))  _Float16 v8h;
typedef __attribute__((ext_vector_type(16))) _Float16 v16h;
typedef __attribute__((ext_vector_type(8)))  float    v8f;
typedef int v4i __attribute__((vector_size(16)));

#if defined(__HIP_DEVICE_COMPILE__) && __has_builtin(__builtin_amdgcn_global_load_async_to_lds_b128)
#define PTR_ASYNC_LDS 1
#else
#define PTR_ASYNC_LDS 0
#endif

#if PTR_ASYNC_LDS
__device__ __forceinline__ void async_copy_b128(const void* g, void* l) {
  __builtin_amdgcn_global_load_async_to_lds_b128(
      (__attribute__((address_space(1))) v4i*)g,
      (__attribute__((address_space(3))) v4i*)l, 0, 0);
}
__device__ __forceinline__ void async_wait_0() {
#if __has_builtin(__builtin_amdgcn_s_wait_asynccnt)
  __builtin_amdgcn_s_wait_asynccnt(0);
#else
  asm volatile("s_wait_asynccnt 0x0" ::: "memory");
#endif
}
__device__ __forceinline__ void async_wait_1() {
#if __has_builtin(__builtin_amdgcn_s_wait_asynccnt)
  __builtin_amdgcn_s_wait_asynccnt(1);
#else
  asm volatile("s_wait_asynccnt 0x1" ::: "memory");
#endif
}
#endif

__device__ __forceinline__ float sigf(float x) { return 1.0f / (1.0f + __expf(-x)); }

__device__ __forceinline__ v8f wmma16(v16h a, v16h b, v8f c) {
  // D = A(16x32 f16) * B(32x16 f16) + C(16x16 f32)
  return __builtin_amdgcn_wmma_f32_16x16x32_f16(false, a, false, b, (short)0, c,
                                                false, false);
}

// A-fragment (16x32 f16) from a row-major f16 row in LDS.
// Lane m = lane&15 holds row M=m; hi = lane>>4 selects K ranges
// {hi*8..hi*8+7, 16+hi*8..16+hi*8+7} -> two ds_load_b128.
__device__ __forceinline__ v16h load_a_frag(const _Float16* row, int kbase, int hi) {
  v8h lo = *(const v8h*)(row + kbase + hi * 8);
  v8h hv = *(const v8h*)(row + kbase + 16 + hi * 8);
  v16h a;
#pragma unroll
  for (int i = 0; i < 8; ++i) { a[i] = lo[i]; a[8 + i] = hv[i]; }
  return a;
}

// ---------------------------------------------------------------------------
// Pack W (f32, [N,K] row-major; computes y = x @ W^T) into B-fragment layout:
// frag (kt,jt), lane l, half i  <-  W[jt*16 + (l&15)][kt*32 + (i>>3)*16 + (l>>4)*8 + (i&7)]
// ---------------------------------------------------------------------------
__global__ void ptr_pack_b_kernel(const float* __restrict__ W,
                                  _Float16* __restrict__ out, int N, int K) {
  int tid = blockIdx.x * blockDim.x + threadIdx.x;
  if (tid >= N * K) return;
  int i    = tid & 15;
  int lane = (tid >> 4) & 31;
  int frag = tid >> 9;
  int NT = N >> 4;
  int jt = frag % NT;
  int kt = frag / NT;
  int k = kt * 32 + ((i >> 3) << 4) + ((lane >> 4) << 3) + (i & 7);
  int n = jt * 16 + (lane & 15);
  out[tid] = (_Float16)W[(size_t)n * K + k];
}

// x[(s*B+b)*E + e] = emb[inputs[b*S+s]][e]
__global__ void ptr_embed_kernel(const int* __restrict__ inputs,
                                 const float* __restrict__ emb,
                                 _Float16* __restrict__ x, int S, int B, int E) {
  int tid = blockIdx.x * blockDim.x + threadIdx.x;
  if (tid >= S * B * E) return;
  int e = tid % E;
  int sb = tid / E;
  int b = sb % B;
  int s = sb / B;
  x[tid] = (_Float16)emb[(size_t)inputs[b * S + s] * E + e];
}

// gx0row[j] = d0_b[j] + sum_k emb[0][k] * d0_Wi[j][k]   (constant decoder input)
__global__ void ptr_gx0row_kernel(const float* __restrict__ emb0,
                                  const float* __restrict__ Wi,
                                  const float* __restrict__ bias,
                                  float* __restrict__ out, int K, int N) {
  int j = blockIdx.x * blockDim.x + threadIdx.x;
  if (j >= N) return;
  float acc = bias[j];
  for (int k = 0; k < K; ++k) acc += emb0[k] * Wi[(size_t)j * K + k];
  out[j] = acc;
}

__global__ void ptr_add2_kernel(const float* __restrict__ a,
                                const float* __restrict__ b,
                                float* __restrict__ o, int n) {
  int tid = blockIdx.x * blockDim.x + threadIdx.x;
  if (tid < n) o[tid] = a[tid] + b[tid];
}

// ---------------------------------------------------------------------------
// Generic WMMA GEMM: out_f16[M,N] = A_f16[M,K] @ Bpacked + bias.
// Grid (M/16, N/128), 256 threads = 8 waves, one 16x16 tile per wave.
// A staged in LDS; async path double-buffers the staging (issue k+1, wait k).
// ---------------------------------------------------------------------------
#define GEMM_KC 128
__global__ void ptr_gemm16_kernel(const _Float16* __restrict__ A,
                                  const _Float16* __restrict__ Bp,
                                  const float* __restrict__ bias,
                                  _Float16* __restrict__ out,
                                  int M, int K, int N) {
  __shared__ _Float16 At[2][16][GEMM_KC + 8];
  int tid = threadIdx.x, lane = tid & 31, wave = tid >> 5;
  int mt = blockIdx.x;
  int jt = blockIdx.y * 8 + wave;
  int NT = N >> 4;
  int m0 = mt * 16;
  int n = lane & 15, hi = lane >> 4;

  v8f c;
  float bv = bias[jt * 16 + n];
#pragma unroll
  for (int r = 0; r < 8; ++r) c[r] = bv;

  int nchunks = K / GEMM_KC;
  int lrow = tid >> 4, lcol = (tid & 15) << 3;
  const _Float16* arow = &A[(size_t)(m0 + lrow) * K + lcol];

#if PTR_ASYNC_LDS
  async_copy_b128(arow, &At[0][lrow][lcol]);  // preload chunk 0
#endif
  for (int kc = 0; kc < nchunks; ++kc) {
#if PTR_ASYNC_LDS
    __syncthreads();  // everyone done reading buffer (kc+1)&1 from step kc-1
    if (kc + 1 < nchunks) {
      async_copy_b128(arow + (kc + 1) * GEMM_KC, &At[(kc + 1) & 1][lrow][lcol]);
      async_wait_1();  // chunk kc landed (async loads complete in order)
    } else {
      async_wait_0();
    }
    __syncthreads();
    const _Float16(*Atile)[GEMM_KC + 8] = At[kc & 1];
#else
    __syncthreads();
    *(v8h*)&At[0][lrow][lcol] = *(const v8h*)(arow + kc * GEMM_KC);
    __syncthreads();
    const _Float16(*Atile)[GEMM_KC + 8] = At[0];
#endif
    if (kc + 1 < nchunks)  // global_prefetch_b8 on next weight chunk
      __builtin_prefetch(&Bp[((size_t)((kc + 1) * (GEMM_KC / 32) * NT + jt) * 32 + lane) * 16], 0, 1);
#pragma unroll
    for (int q = 0; q < GEMM_KC / 32; ++q) {
      int kt = kc * (GEMM_KC / 32) + q;
      v16h a = load_a_frag(&Atile[n][0], q * 32, hi);
      v16h b = *(const v16h*)&Bp[((size_t)(kt * NT + jt) * 32 + lane) * 16];
      c = wmma16(a, b, c);
    }
  }
#pragma unroll
  for (int r = 0; r < 8; ++r) {
    int mrow = m0 + (hi ? r + 8 : r);
    out[(size_t)mrow * N + jt * 16 + n] = (_Float16)c[r];
  }
}

// ---------------------------------------------------------------------------
// Persistent LSTM scan. One WG per 16 batch rows; h (f16) / c (f32) in LDS.
// Per step: async-stage GX[s] tile (64KB, contiguous) into LDS, overlap with
// recurrent WMMA (zero-init accumulator), wait, then pointwise cell adding
// the hoisted gates coalesced. Dynamic LDS = 245760 B.
// ---------------------------------------------------------------------------
__global__ void ptr_lstm_scan_kernel(const _Float16* __restrict__ GX,
                                     const _Float16* __restrict__ WhP,
                                     _Float16* __restrict__ out,
                                     int out_ss, int out_bs, int out_off,
                                     float* __restrict__ hfin, float* __restrict__ cfin,
                                     int S, int B, int reverse, int accumulate) {
  extern __shared__ char smem[];
  float*    gS  = (float*)smem;                 // [16][2048] f32  131072
  _Float16* hS  = (_Float16*)(smem + 131072);   // [16][512]  f16   16384
  float*    cS  = (float*)(smem + 147456);      // [16][512]  f32   32768
  _Float16* gxS = (_Float16*)(smem + 180224);   // [16][2048] f16   65536 (async)

  int tid = threadIdx.x, lane = tid & 31, wave = tid >> 5;
  int b0 = blockIdx.x * 16;
  int n = lane & 15, hi = lane >> 4;

  for (int idx = tid; idx < 16 * 512; idx += 256) { hS[idx] = (_Float16)0.0f; cS[idx] = 0.0f; }
  __syncthreads();

  for (int t = 0; t < S; ++t) {
    int s = reverse ? (S - 1 - t) : t;
    const _Float16* gsrc = &GX[((size_t)s * B + b0) * 2048];
#if PTR_ASYNC_LDS
    // stage this step's hoisted gates; rides under the WMMA below
#pragma unroll
    for (int i = 0; i < 16; ++i) {
      int o = tid * 256 + i * 16;
      async_copy_b128((const char*)gsrc + o, (char*)gxS + o);
    }
#endif
    for (int u = 0; u < 16; ++u) {  // 8 waves x 16 = 128 j-tiles
      int jt = wave + 8 * u;
      v8f c = {};
#pragma unroll
      for (int kt = 0; kt < 16; ++kt) {  // K = 512 recurrent GEMM
        v16h a  = load_a_frag(&hS[n * 512], kt * 32, hi);
        v16h bf = *(const v16h*)&WhP[((size_t)(kt * 128 + jt) * 32 + lane) * 16];
        c = wmma16(a, bf, c);
      }
#pragma unroll
      for (int r = 0; r < 8; ++r) gS[(hi * 8 + r) * 2048 + jt * 16 + n] = c[r];
    }
#if PTR_ASYNC_LDS
    async_wait_0();
#endif
    __syncthreads();
    for (int idx = tid; idx < 16 * 512; idx += 256) {
      int r = idx >> 9, h = idx & 511;
#if PTR_ASYNC_LDS
      const _Float16* gx = &gxS[r * 2048];
#else
      const _Float16* gx = gsrc + (size_t)r * 2048;
#endif
      float gi = gS[r * 2048 + h]        + (float)gx[h];
      float gf = gS[r * 2048 + 512 + h]  + (float)gx[512 + h];
      float gg = gS[r * 2048 + 1024 + h] + (float)gx[1024 + h];
      float go = gS[r * 2048 + 1536 + h] + (float)gx[1536 + h];
      float cc = sigf(gf) * cS[idx] + sigf(gi) * tanhf(gg);
      float hh = sigf(go) * tanhf(cc);
      cS[idx] = cc;
      hS[idx] = (_Float16)hh;
      size_t oi = (size_t)s * out_ss + (size_t)(b0 + r) * out_bs + out_off + h;
      if (accumulate) out[oi] = (_Float16)((float)out[oi] + hh);
      else            out[oi] = (_Float16)hh;
    }
    __syncthreads();
  }
  for (int idx = tid; idx < 16 * 512; idx += 256) {
    int r = idx >> 9, h = idx & 511;
    hfin[(size_t)(b0 + r) * 512 + h] = (float)hS[idx];
    cfin[(size_t)(b0 + r) * 512 + h] = cS[idx];
  }
}

// ---------------------------------------------------------------------------
// Persistent decoder: 1 WG per 16 batch rows, 128 steps.
// Constant vectors (gx0row, d1_b, b2, v) staged in LDS once.
// Dynamic LDS = 283648 B.
// ---------------------------------------------------------------------------
__global__ void ptr_decoder_kernel(const float* __restrict__ gx0row,
                                   const _Float16* __restrict__ d0WhP,
                                   const _Float16* __restrict__ d1WiP,
                                   const _Float16* __restrict__ d1WhP,
                                   const float* __restrict__ d1b,
                                   const _Float16* __restrict__ w2P,
                                   const float* __restrict__ b2,
                                   const float* __restrict__ vvec,
                                   const _Float16* __restrict__ encE,
                                   const float* __restrict__ h0i, const float* __restrict__ c0i,
                                   const float* __restrict__ h1i, const float* __restrict__ c1i,
                                   float* __restrict__ out, int S, int B) {
  extern __shared__ char smem[];
  float*    gS   = (float*)smem;                 // 131072
  _Float16* h0S  = (_Float16*)(smem + 131072);   //  16384
  _Float16* h1S  = (_Float16*)(smem + 147456);   //  16384
  float*    c0S  = (float*)(smem + 163840);      //  32768
  float*    c1S  = (float*)(smem + 196608);      //  32768
  float*    deS  = (float*)(smem + 229376);      //  32768
  float*    redS = (float*)(smem + 262144);      //   1024
  float*    vS   = (float*)(smem + 263168);      //   2048
  float*    gx0S = (float*)(smem + 265216);      //   8192
  float*    d1bS = (float*)(smem + 273408);      //   8192
  float*    b2S  = (float*)(smem + 281600);      //   2048

  int tid = threadIdx.x, lane = tid & 31, wave = tid >> 5;
  int b0 = blockIdx.x * 16;
  int n = lane & 15, hi = lane >> 4;

  for (int idx = tid; idx < 16 * 512; idx += 256) {
    int r = idx >> 9, h = idx & 511;
    h0S[idx] = (_Float16)h0i[(size_t)(b0 + r) * 512 + h];
    h1S[idx] = (_Float16)h1i[(size_t)(b0 + r) * 512 + h];
    c0S[idx] = c0i[(size_t)(b0 + r) * 512 + h];
    c1S[idx] = c1i[(size_t)(b0 + r) * 512 + h];
  }
  for (int idx = tid; idx < 2048; idx += 256) { gx0S[idx] = gx0row[idx]; d1bS[idx] = d1b[idx]; }
  for (int idx = tid; idx < 512; idx += 256)  { vS[idx] = vvec[idx]; b2S[idx] = b2[idx]; }
  __syncthreads();

  for (int step = 0; step < S; ++step) {
    // ---------- d0 cell: g = gx0 (const input term + bias) + h0 @ d0Wh^T
    for (int u = 0; u < 16; ++u) {
      int jt = wave + 8 * u;
      v8f c;
      float base = gx0S[jt * 16 + n];
#pragma unroll
      for (int r = 0; r < 8; ++r) c[r] = base;
#pragma unroll
      for (int kt = 0; kt < 16; ++kt) {
        v16h a  = load_a_frag(&h0S[n * 512], kt * 32, hi);
        v16h bf = *(const v16h*)&d0WhP[((size_t)(kt * 128 + jt) * 32 + lane) * 16];
        c = wmma16(a, bf, c);
      }
#pragma unroll
      for (int r = 0; r < 8; ++r) gS[(hi * 8 + r) * 2048 + jt * 16 + n] = c[r];
    }
    __syncthreads();
    for (int idx = tid; idx < 16 * 512; idx += 256) {
      int r = idx >> 9, h = idx & 511;
      float gi = gS[r * 2048 + h], gf = gS[r * 2048 + 512 + h];
      float gg = gS[r * 2048 + 1024 + h], go = gS[r * 2048 + 1536 + h];
      float cc = sigf(gf) * c0S[idx] + sigf(gi) * tanhf(gg);
      c0S[idx] = cc;
      h0S[idx] = (_Float16)(sigf(go) * tanhf(cc));
    }
    __syncthreads();
    // ---------- d1 cell: g = d1_b + h0 @ d1Wi^T + h1 @ d1Wh^T
    for (int u = 0; u < 16; ++u) {
      int jt = wave + 8 * u;
      v8f c;
      float base = d1bS[jt * 16 + n];
#pragma unroll
      for (int r = 0; r < 8; ++r) c[r] = base;
#pragma unroll
      for (int kt = 0; kt < 16; ++kt) {
        v16h a  = load_a_frag(&h0S[n * 512], kt * 32, hi);
        v16h bf = *(const v16h*)&d1WiP[((size_t)(kt * 128 + jt) * 32 + lane) * 16];
        c = wmma16(a, bf, c);
      }
#pragma unroll
      for (int kt = 0; kt < 16; ++kt) {
        v16h a  = load_a_frag(&h1S[n * 512], kt * 32, hi);
        v16h bf = *(const v16h*)&d1WhP[((size_t)(kt * 128 + jt) * 32 + lane) * 16];
        c = wmma16(a, bf, c);
      }
#pragma unroll
      for (int r = 0; r < 8; ++r) gS[(hi * 8 + r) * 2048 + jt * 16 + n] = c[r];
    }
    __syncthreads();
    for (int idx = tid; idx < 16 * 512; idx += 256) {
      int r = idx >> 9, h = idx & 511;
      float gi = gS[r * 2048 + h], gf = gS[r * 2048 + 512 + h];
      float gg = gS[r * 2048 + 1024 + h], go = gS[r * 2048 + 1536 + h];
      float cc = sigf(gf) * c1S[idx] + sigf(gi) * tanhf(gg);
      c1S[idx] = cc;
      h1S[idx] = (_Float16)(sigf(go) * tanhf(cc));
    }
    __syncthreads();
    // ---------- de = b2 + h1 @ w2^T  (N = 512 -> 32 j-tiles)
    for (int u = 0; u < 4; ++u) {
      int jt = wave + 8 * u;
      v8f c;
      float base = b2S[jt * 16 + n];
#pragma unroll
      for (int r = 0; r < 8; ++r) c[r] = base;
#pragma unroll
      for (int kt = 0; kt < 16; ++kt) {
        v16h a  = load_a_frag(&h1S[n * 512], kt * 32, hi);
        v16h bf = *(const v16h*)&w2P[((size_t)(kt * 32 + jt) * 32 + lane) * 16];
        c = wmma16(a, bf, c);
      }
#pragma unroll
      for (int r = 0; r < 8; ++r) deS[(hi * 8 + r) * 512 + jt * 16 + n] = c[r];
    }
    __syncthreads();
    // ---------- attention + softmax (16 threads per batch row)
    int r = tid >> 4, l16 = tid & 15;
    float evals[8];
    float emax = -1e30f;
#pragma unroll
    for (int i = 0; i < 8; ++i) {
      int s = l16 * 8 + i;
      const _Float16* ep = &encE[((size_t)(b0 + r) * S + s) * 512];
      float acc = 0.0f;
      for (int h = 0; h < 512; ++h)
        acc += tanhf((float)ep[h] + deS[r * 512 + h]) * vS[h];
      evals[i] = acc;
      emax = fmaxf(emax, acc);
    }
    redS[r * 16 + l16] = emax;
    __syncthreads();
    if (l16 == 0) {
      float m = redS[r * 16];
      for (int q = 1; q < 16; ++q) m = fmaxf(m, redS[r * 16 + q]);
      redS[r * 16] = m;
    }
    __syncthreads();
    float m = redS[r * 16];
    float psum = 0.0f;
#pragma unroll
    for (int i = 0; i < 8; ++i) { evals[i] = __expf(evals[i] - m); psum += evals[i]; }
    __syncthreads();
    redS[r * 16 + l16] = psum;
    __syncthreads();
    if (l16 == 0) {
      float ssum = 0.0f;
      for (int q = 0; q < 16; ++q) ssum += redS[r * 16 + q];
      redS[r * 16] = ssum;
    }
    __syncthreads();
    float inv = 1.0f / redS[r * 16];
#pragma unroll
    for (int i = 0; i < 8; ++i) {
      int s = l16 * 8 + i;
      out[((size_t)(b0 + r) * S + s) * S + step] = evals[i] * inv;
    }
    __syncthreads();
  }
}

// ---------------------------------------------------------------------------
// Host orchestration
// ---------------------------------------------------------------------------
extern "C" void kernel_launch(void* const* d_in, const int* in_sizes, int n_in,
                              void* d_out, int out_size, void* d_ws, size_t ws_size,
                              hipStream_t stream) {
  (void)in_sizes; (void)n_in; (void)out_size; (void)ws_size;
  const int B = 256, S = 128, E = 256, H = 512;
  const int G = 4 * H;  // 2048

  const int*   inputs = (const int*)d_in[0];
  const float* emb    = (const float*)d_in[1];
  const float* e0f_Wi = (const float*)d_in[2];
  const float* e0f_Wh = (const float*)d_in[3];
  const float* e0f_b  = (const float*)d_in[4];
  const float* e0b_Wi = (const float*)d_in[5];
  const float* e0b_Wh = (const float*)d_in[6];
  const float* e0b_b  = (const float*)d_in[7];
  const float* e1f_Wi = (const float*)d_in[8];
  const float* e1f_Wh = (const float*)d_in[9];
  const float* e1f_b  = (const float*)d_in[10];
  const float* e1b_Wi = (const float*)d_in[11];
  const float* e1b_Wh = (const float*)d_in[12];
  const float* e1b_b  = (const float*)d_in[13];
  const float* d0_Wi  = (const float*)d_in[14];
  const float* d0_Wh  = (const float*)d_in[15];
  const float* d0_b   = (const float*)d_in[16];
  const float* d1_Wi  = (const float*)d_in[17];
  const float* d1_Wh  = (const float*)d_in[18];
  const float* d1_b   = (const float*)d_in[19];
  const float* w1     = (const float*)d_in[20];
  const float* b1     = (const float*)d_in[21];
  const float* w2     = (const float*)d_in[22];
  const float* b2     = (const float*)d_in[23];
  const float* vvec   = (const float*)d_in[24];

  char* ws = (char*)d_ws;
  size_t off = 0;
  auto alloc = [&](size_t bytes) -> char* {
    char* p = ws + off;
    off += (bytes + 255) & ~(size_t)255;
    return p;
  };
  auto pack = [&](const float* W, int N, int K) -> _Float16* {
    _Float16* p = (_Float16*)alloc((size_t)N * K * 2);
    ptr_pack_b_kernel<<<(N * K) / 256, 256, 0, stream>>>(W, p, N, K);
    return p;
  };

  _Float16* P_e0fWi = pack(e0f_Wi, G, E);
  _Float16* P_e0fWh = pack(e0f_Wh, G, H);
  _Float16* P_e0bWi = pack(e0b_Wi, G, E);
  _Float16* P_e0bWh = pack(e0b_Wh, G, H);
  _Float16* P_e1fWi = pack(e1f_Wi, G, 2 * H);
  _Float16* P_e1fWh = pack(e1f_Wh, G, H);
  _Float16* P_e1bWi = pack(e1b_Wi, G, 2 * H);
  _Float16* P_e1bWh = pack(e1b_Wh, G, H);
  _Float16* P_d0Wh  = pack(d0_Wh, G, H);
  _Float16* P_d1Wi  = pack(d1_Wi, G, H);
  _Float16* P_d1Wh  = pack(d1_Wh, G, H);
  _Float16* P_w1    = pack(w1, H, H);
  _Float16* P_w2    = pack(w2, H, H);

  _Float16* xemb    = (_Float16*)alloc((size_t)S * B * E * 2);      // [s][b][E]
  _Float16* xcat    = (_Float16*)alloc((size_t)S * B * 2 * H * 2);  // [s][b][2H]
  _Float16* enc_sum = (_Float16*)alloc((size_t)B * S * H * 2);      // [b][s][H]
  _Float16* encE    = (_Float16*)alloc((size_t)B * S * H * 2);      // [b][s][H]
  _Float16* GX      = (_Float16*)alloc((size_t)S * B * G * 2);      // reused 4x
  float* hf0 = (float*)alloc((size_t)B * H * 4);
  float* cf0 = (float*)alloc((size_t)B * H * 4);
  float* hb0 = (float*)alloc((size_t)B * H * 4);
  float* cb0 = (float*)alloc((size_t)B * H * 4);
  float* hf1 = (float*)alloc((size_t)B * H * 4);
  float* cf1 = (float*)alloc((size_t)B * H * 4);
  float* hb1 = (float*)alloc((size_t)B * H * 4);
  float* cb1 = (float*)alloc((size_t)B * H * 4);
  float* h0s = (float*)alloc((size_t)B * H * 4);
  float* c0s = (float*)alloc((size_t)B * H * 4);
  float* h1s = (float*)alloc((size_t)B * H * 4);
  float* c1s = (float*)alloc((size_t)B * H * 4);
  float* gx0 = (float*)alloc((size_t)G * 4);

  const size_t LSTM_LDS = 245760;  // gS + hS + cS + gxS (async staging)
  const size_t DEC_LDS  = 283648;
  const int M = S * B;  // 32768

  ptr_embed_kernel<<<(S * B * E) / 256, 256, 0, stream>>>(inputs, emb, xemb, S, B, E);

  // ---- encoder layer 0 ----
  ptr_gemm16_kernel<<<dim3(M / 16, G / 128), 256, 0, stream>>>(xemb, P_e0fWi, e0f_b, GX, M, E, G);
  ptr_lstm_scan_kernel<<<16, 256, LSTM_LDS, stream>>>(GX, P_e0fWh, xcat, B * 2 * H, 2 * H, 0,
                                                      hf0, cf0, S, B, 0, 0);
  ptr_gemm16_kernel<<<dim3(M / 16, G / 128), 256, 0, stream>>>(xemb, P_e0bWi, e0b_b, GX, M, E, G);
  ptr_lstm_scan_kernel<<<16, 256, LSTM_LDS, stream>>>(GX, P_e0bWh, xcat, B * 2 * H, 2 * H, H,
                                                      hb0, cb0, S, B, 1, 0);
  // ---- encoder layer 1 (fwd writes, bwd accumulates into enc_sum) ----
  ptr_gemm16_kernel<<<dim3(M / 16, G / 128), 256, 0, stream>>>(xcat, P_e1fWi, e1f_b, GX, M, 2 * H, G);
  ptr_lstm_scan_kernel<<<16, 256, LSTM_LDS, stream>>>(GX, P_e1fWh, enc_sum, H, S * H, 0,
                                                      hf1, cf1, S, B, 0, 0);
  ptr_gemm16_kernel<<<dim3(M / 16, G / 128), 256, 0, stream>>>(xcat, P_e1bWi, e1b_b, GX, M, 2 * H, G);
  ptr_lstm_scan_kernel<<<16, 256, LSTM_LDS, stream>>>(GX, P_e1bWh, enc_sum, H, S * H, 0,
                                                      hb1, cb1, S, B, 1, 1);

  // decoder initial states: sum of directions per layer
  int nf = B * H;
  ptr_add2_kernel<<<nf / 256, 256, 0, stream>>>(hf0, hb0, h0s, nf);
  ptr_add2_kernel<<<nf / 256, 256, 0, stream>>>(cf0, cb0, c0s, nf);
  ptr_add2_kernel<<<nf / 256, 256, 0, stream>>>(hf1, hb1, h1s, nf);
  ptr_add2_kernel<<<nf / 256, 256, 0, stream>>>(cf1, cb1, c1s, nf);

  // constant decoder-input gate term: gx0 = emb[0] @ d0_Wi^T + d0_b
  ptr_gx0row_kernel<<<G / 256, 256, 0, stream>>>(emb, d0_Wi, d0_b, gx0, E, G);

  // enc_energy = enc_sum @ w1^T + b1  ([B*S, H] x [H, H])
  ptr_gemm16_kernel<<<dim3(M / 16, H / 128), 256, 0, stream>>>(enc_sum, P_w1, b1, encE, M, H, H);

  // decoder scan + attention
  ptr_decoder_kernel<<<16, 256, DEC_LDS, stream>>>(gx0, P_d0Wh, P_d1Wi, P_d1Wh, d1_b,
                                                   P_w2, b2, vvec, encE,
                                                   h0s, c0s, h1s, c1s,
                                                   (float*)d_out, S, B);
}
